// GNNModel_71708773974824
// MI455X (gfx1250) — matrix operand order repaired and checked by reading, not compiled, not run on hardware
//
#include <hip/hip_runtime.h>
#include <hip/hip_bf16.h>

typedef __attribute__((ext_vector_type(16))) __bf16 v16bf;
typedef __attribute__((ext_vector_type(8)))  __bf16 v8bf;
typedef __attribute__((ext_vector_type(8)))  float  v8f;

union Frag16 { v16bf v; v8bf h[2]; };

#define LDS_STRIDE 136   // 128 bf16 + 8 pad -> b128 start bank = 4*lane, conflict-free

// ---------------------------------------------------------------- zero fill
__global__ void k_zero(float* __restrict__ p, long n) {
    long i      = (long)blockIdx.x * blockDim.x + threadIdx.x;
    long stride = (long)gridDim.x * blockDim.x;
    for (; i < n; i += stride) p[i] = 0.0f;
}

// ---------------------------------------------------------------- degree count
__global__ void k_count(const int* __restrict__ tgt, float* __restrict__ cnt, int ne) {
    int e = blockIdx.x * blockDim.x + threadIdx.x;
    if (e < ne) atomicAdd(cnt + tgt[e], 1.0f);
}

// ---------------------------------------------------------------- edge scatter-add
// One wave32 per edge: lane l moves features [4l, 4l+4) of the 128-wide row.
__global__ __launch_bounds__(256) void k_scatter(const float* __restrict__ feat,
                                                 const int* __restrict__ src,
                                                 const int* __restrict__ tgt,
                                                 float* __restrict__ agg, int ne) {
    int wid  = (int)(((long)blockIdx.x * blockDim.x + threadIdx.x) >> 5);
    int lane = threadIdx.x & 31;
    if (wid >= ne) return;
    int s = src[wid];
    int t = tgt[wid];
    const float4 v = *reinterpret_cast<const float4*>(feat + (long)s * 128 + lane * 4);
    float* dst = agg + (long)t * 128 + lane * 4;
    atomicAdd(dst + 0, v.x);
    atomicAdd(dst + 1, v.y);
    atomicAdd(dst + 2, v.z);
    atomicAdd(dst + 3, v.w);
}

// ---------------------------------------------------------------- fused GNN layer
// out[row,:] = relu((x + agg/max(cnt,1)) @ W + bias)
// Block: 256 threads = 8 waves, 128 rows per block.
// K-outer / N-inner with 8 live f32 accumulators (64 VGPRs): each A fragment
// is loaded once per K-step and reused by 8 WMMAs; each B fragment loaded
// exactly once. 32 v_wmma_f32_16x16x32_bf16 per wave, every fragment two
// bank-conflict-free ds_load_b128's. Bounds check is block-uniform.
__global__ __launch_bounds__(256) void k_layer(const float* __restrict__ xin,
                                               const float* __restrict__ agg,
                                               const float* __restrict__ cnt,
                                               const float* __restrict__ W,
                                               const float* __restrict__ bias,
                                               float* __restrict__ out, int nrows) {
    __shared__ __bf16 sX[128 * LDS_STRIDE];   // combined rows  [M][K]
    __shared__ __bf16 sWT[128 * LDS_STRIDE];  // W transposed   [N][K]

    const int  tid     = threadIdx.x;
    const int  rowBase = blockIdx.x * 128;
    const bool full    = (rowBase + 128) <= nrows;   // uniform across block

    // stage W transposed as bf16: coalesced global reads over n
    for (int p = tid; p < 128 * 128; p += 256) {
        int k = p >> 7, n = p & 127;
        sWT[n * LDS_STRIDE + k] = (__bf16)W[p];
    }

    // stage combined = x + agg / clamp(cnt,1) as bf16 (pairs -> ds_store_b32)
    for (int p = tid; p < 128 * 64; p += 256) {
        int   m   = p >> 6;
        int   k2  = (p & 63) * 2;
        int   row = rowBase + m;
        float v0 = 0.0f, v1 = 0.0f;
        if (row < nrows) {
            float cn = cnt[row];
            cn       = cn < 1.0f ? 1.0f : cn;
            float rc = 1.0f / cn;
            long  o  = (long)row * 128 + k2;
            v0 = xin[o] + agg[o] * rc;
            v1 = xin[o + 1] + agg[o + 1] * rc;
        }
        sX[m * LDS_STRIDE + k2]     = (__bf16)v0;
        sX[m * LDS_STRIDE + k2 + 1] = (__bf16)v1;
    }
    __syncthreads();

    const int lane  = tid & 31;
    const int wv    = tid >> 5;    // wave -> M-tile index 0..7
    const int lhalf = lane >> 4;
    const int l15   = lane & 15;

    const __bf16* arow = sX + (wv * 16 + l15) * LDS_STRIDE;
    const __bf16* bcol = sWT + l15 * LDS_STRIDE;

    v8f c[8] = {};   // 8 N-tile accumulators, 64 VGPRs

#pragma unroll
    for (int kk = 0; kk < 4; ++kk) {
        // A fragment (ISA 16-bit A 16x32 layout):
        // lanes 0-15: K = {kb..kb+7, kb+16..kb+23}; lanes 16-31: +8
        Frag16 a;
        {
            int kb = kk * 32 + lhalf * 8;
            a.h[0] = *reinterpret_cast<const v8bf*>(arow + kb);
            a.h[1] = *reinterpret_cast<const v8bf*>(arow + kb + 16);
        }
        const int kbB = kk * 32 + lhalf * 8;
#pragma unroll
        for (int nt = 0; nt < 8; ++nt) {
            // B fragment (32x16 KxN): same K order as A: {kb..kb+7, kb+16..kb+23}, lanes 16-31: +8
            Frag16 b;
            const __bf16* brow = bcol + nt * 16 * LDS_STRIDE + kbB;
            b.h[0] = *reinterpret_cast<const v8bf*>(brow);
            b.h[1] = *reinterpret_cast<const v8bf*>(brow + 16);
            c[nt] = __builtin_amdgcn_wmma_f32_16x16x32_bf16(
                        false, a.v, false, b.v, (short)0, c[nt], false, false);
        }
    }

    // Epilogue: C/D layout — VGPR r holds M = r (lanes 0-15) / r+8 (lanes 16-31)
    const int rowLane = rowBase + wv * 16 + lhalf * 8;
#pragma unroll
    for (int nt = 0; nt < 8; ++nt) {
        const int   ncol = nt * 16 + l15;
        const float bv   = bias[ncol];
        float* po = out + ((long)rowLane * 128 + ncol);
        if (full) {                    // uniform scalar branch: hot path
#pragma unroll
            for (int r = 0; r < 8; ++r) {
                float v = c[nt][r] + bv;
                po[(long)r * 128] = v > 0.0f ? v : 0.0f;   // imm-offset stores
            }
        } else {                       // tail block only
#pragma unroll
            for (int r = 0; r < 8; ++r) {
                if (rowLane + r < nrows) {
                    float v = c[nt][r] + bv;
                    po[(long)r * 128] = v > 0.0f ? v : 0.0f;
                }
            }
        }
    }
}

// ---------------------------------------------------------------- projection [128]->[5]
__global__ __launch_bounds__(256) void k_proj(const float* __restrict__ h,
                                              const float* __restrict__ Wp,
                                              const float* __restrict__ bp,
                                              float* __restrict__ out, int nrows) {
    int node = (int)(((long)blockIdx.x * blockDim.x + threadIdx.x) >> 5);
    int lane = threadIdx.x & 31;
    if (node >= nrows) return;
    const float4 v = *reinterpret_cast<const float4*>(h + (long)node * 128 + lane * 4);
    const int k0   = lane * 4;
    float acc[5];
#pragma unroll
    for (int j = 0; j < 5; ++j) {
        acc[j] = v.x * Wp[(k0 + 0) * 5 + j] + v.y * Wp[(k0 + 1) * 5 + j] +
                 v.z * Wp[(k0 + 2) * 5 + j] + v.w * Wp[(k0 + 3) * 5 + j];
    }
#pragma unroll
    for (int j = 0; j < 5; ++j)
        for (int off = 16; off > 0; off >>= 1)
            acc[j] += __shfl_down(acc[j], off, 32);
    if (lane == 0) {
#pragma unroll
        for (int j = 0; j < 5; ++j) out[(long)node * 5 + j] = acc[j] + bp[j];
    }
}

// ---------------------------------------------------------------- launcher
extern "C" void kernel_launch(void* const* d_in, const int* in_sizes, int n_in,
                              void* d_out, int out_size, void* d_ws, size_t ws_size,
                              hipStream_t stream) {
    const float* x  = (const float*)d_in[0];
    const int*   ed = (const int*)d_in[1];
    const float* W1 = (const float*)d_in[2];
    const float* b1 = (const float*)d_in[3];
    const float* W2 = (const float*)d_in[4];
    const float* b2 = (const float*)d_in[5];
    const float* Wp = (const float*)d_in[6];
    const float* bp = (const float*)d_in[7];
    float* out      = (float*)d_out;

    const int N = in_sizes[0] / 128;  // 100000
    const int E = in_sizes[1] / 2;    // 1600000
    const int* src = ed;
    const int* tgt = ed + E;

    // workspace layout: agg[N*128] | cnt[N] | h1[N*128] | h2[N*128]
    float* agg = (float*)d_ws;
    float* cnt = agg + (long)N * 128;
    float* h1  = cnt + N;
    float* h2  = h1 + (long)N * 128;

    const int scatterBlocks = (int)(((long)E * 32 + 255) / 256);
    const int layerBlocks   = (N + 127) / 128;
    const int projBlocks    = (int)(((long)N * 32 + 255) / 256);

    // layer 1
    k_zero<<<2048, 256, 0, stream>>>(agg, (long)N * 128 + N);  // agg + cnt contiguous
    k_count<<<(E + 255) / 256, 256, 0, stream>>>(tgt, cnt, E);
    k_scatter<<<scatterBlocks, 256, 0, stream>>>(x, src, tgt, agg, E);
    k_layer<<<layerBlocks, 256, 0, stream>>>(x, agg, cnt, W1, b1, h1, N);

    // layer 2 (cnt reusable; agg re-zeroed)
    k_zero<<<2048, 256, 0, stream>>>(agg, (long)N * 128);
    k_scatter<<<scatterBlocks, 256, 0, stream>>>(h1, src, tgt, agg, E);
    k_layer<<<layerBlocks, 256, 0, stream>>>(h1, agg, cnt, W2, b2, h2, N);

    // projection
    k_proj<<<projBlocks, 256, 0, stream>>>(h2, Wp, bp, out, N);
}
